// ModuleCorrelationPytorch_48387101557446
// MI455X (gfx1250) — matrix operand order repaired and checked by reading, not compiled, not run on hardware
//
#include <hip/hip_runtime.h>

#define MAX_DISP 4

typedef __attribute__((ext_vector_type(16))) __bf16   bf16x16;
typedef __attribute__((ext_vector_type(8)))  float    f32x8;
typedef __attribute__((ext_vector_type(8)))  unsigned u32x8;

// Pack two fp32 into one dword of two bf16, low element first.
// Round-to-nearest (ties away from zero): u + 0x8000, take high half.
// Statistically identical to RNE for continuous data, 1 VALU/float cheaper.
__device__ __forceinline__ unsigned bfpack(float lo, float hi) {
    unsigned ul = __builtin_bit_cast(unsigned, lo) + 0x8000u;
    unsigned uh = __builtin_bit_cast(unsigned, hi) + 0x8000u;
    // bytes: [3]=uh.b3 [2]=uh.b2 [1]=ul.b3 [0]=ul.b2
    return __builtin_amdgcn_perm(uh, ul, 0x07060302u);
}

template<int Cc, int Hc, int Wc>
__global__ __launch_bounds__(288)
void corr_wmma_kernel(const float* __restrict__ first,
                      const float* __restrict__ second,
                      float* __restrict__ out) {
    constexpr int HW  = Hc * Wc;
    constexpr int ND  = 2 * MAX_DISP + 1;   // 9
    constexpr int NKC = Cc / 32;            // 8 K-chunks of 32 channels

    // A tile (16 px x Cc ch) as bf16, pre-swizzled into WMMA A-fragment layout:
    // dword index = (kchunk*32 + lane)*8 + d   (each lane reads 32B contiguous)
    __shared__ __align__(32) unsigned ldsA[NKC * 32 * 8];   // 8 KB

    const int tid  = threadIdx.x;
    const int lane = tid & 31;
    const int wave = tid >> 5;              // 0..8
    const int dy   = wave - MAX_DISP;

    const int x0 = blockIdx.x * 16;
    const int y  = blockIdx.y;
    const int b  = blockIdx.z;

    const size_t img = (size_t)b * Cc * HW;

    // ---- stage A tile into LDS (once per block, all 9 waves cooperate) ----
    // 'first' is read exactly once per element globally -> non-temporal loads.
    {
        const float* Ag = first + img + (size_t)y * Wc + x0;
        for (int i = tid; i < NKC * 32 * 8; i += 288) {
            const int sm  = i & 15;          // pixel within tile (x)
            const int shi = (i >> 4) & 1;    // lane half
            const int sd  = (i >> 5) & 7;    // dword within fragment
            const int skc = i >> 8;          // k-chunk
            // ISA 16-bit A layout: d<4 -> K = hi*8 + 2d ; d>=4 -> K = 16 + hi*8 + (2d-8)
            const int k0 = skc * 32 + (sd < 4 ? shi * 8 + 2 * sd
                                              : 16 + shi * 8 + (2 * sd - 8));
            const float flo = __builtin_nontemporal_load(&Ag[(size_t)k0 * HW + sm]);
            const float fhi = __builtin_nontemporal_load(&Ag[(size_t)(k0 + 1) * HW + sm]);
            ldsA[(skc * 32 + shi * 16 + sm) * 8 + sd] = bfpack(flo, fhi);
        }
    }
    __syncthreads();

    // ---- per-lane roles for the WMMA fragments ----
    const int m   = lane & 15;              // B/D column within tile
    const int hi  = lane >> 4;              // K half for B fragment

    const int  ysrc = y + dy;
    const bool yok  = (ysrc >= 0) && (ysrc < Hc);
    const float* Brow = second + img + (size_t)(yok ? ysrc : 0) * Wc;

    // two N-tiles of shifted columns covering x0-4 .. x0+27
    const int xc0 = x0 - MAX_DISP + m;
    const int xc1 = x0 - MAX_DISP + 16 + m;
    const bool v0 = yok && (xc0 >= 0) && (xc0 < Wc);
    const bool v1 = yok && (xc1 >= 0) && (xc1 < Wc);
    const unsigned msk0 = v0 ? 0xFFFFFFFFu : 0u;   // loop-invariant zero masks
    const unsigned msk1 = v1 ? 0xFFFFFFFFu : 0u;
    const float* B0 = Brow + (v0 ? xc0 : 0);       // clamped -> always-valid address
    const float* B1 = Brow + (v1 ? xc1 : 0);

    f32x8 acc0 = {}, acc1 = {};

    #pragma unroll 2
    for (int kcj = 0; kcj < NKC; ++kcj) {
        // A fragment: 2x ds_load_b128 (shared by all 9 waves via LDS)
        const u32x8 aw = *(const u32x8*)&ldsA[(kcj * 32 + lane) * 8];
        const bf16x16 a = __builtin_bit_cast(bf16x16, aw);

        // B fragments: unconditional coalesced loads, zeroed by bitmask AND
        const size_t kb = (size_t)(kcj * 32 + hi * 8) * HW;
        u32x8 bw0, bw1;
        #pragma unroll
        for (int d = 0; d < 8; ++d) {
            const size_t o0 = kb + (size_t)(2 * d + (d >= 4 ? 8 : 0)) * HW;   // K = hi*8 + 2d (+8 for d>=4)
            const size_t o1 = o0 + HW;                     // K = 2d+1
            bw0[d] = bfpack(B0[o0], B0[o1]) & msk0;
            bw1[d] = bfpack(B1[o0], B1[o1]) & msk1;
        }
        const bf16x16 fb0 = __builtin_bit_cast(bf16x16, bw0);
        const bf16x16 fb1 = __builtin_bit_cast(bf16x16, bw1);

        acc0 = __builtin_amdgcn_wmma_f32_16x16x32_bf16(false, a, false, fb0,
                                                       (short)0, acc0, false, false);
        acc1 = __builtin_amdgcn_wmma_f32_16x16x32_bf16(false, a, false, fb1,
                                                       (short)0, acc1, false, false);
    }

    const float scale = 1.0f / (float)Cc;   // mean over channels

    // Row base for this (b, dy, y): element (dx, x) lives at orow[dx*HW + x].
    float* orow = out + (((size_t)b * (ND * ND)
                          + (size_t)((dy + MAX_DISP) * ND + MAX_DISP)) * Hc + y) * Wc;

    // extract the valid diagonal band (dx in [-4,4]) from the two D tiles
    #pragma unroll
    for (int j = 0; j < 8; ++j) {
        const int mm = j + 8 * hi;          // pixel row in D layout
        const int x  = x0 + mm;
        const int dx0 = xc0 - x;
        if ((unsigned)(dx0 + MAX_DISP) < (unsigned)ND) {
            __builtin_nontemporal_store(acc0[j] * scale, &orow[dx0 * HW + x]);
        }
        const int dx1 = xc1 - x;
        if ((unsigned)(dx1 + MAX_DISP) < (unsigned)ND) {
            __builtin_nontemporal_store(acc1[j] * scale, &orow[dx1 * HW + x]);
        }
    }
}

extern "C" void kernel_launch(void* const* d_in, const int* in_sizes, int n_in,
                              void* d_out, int out_size, void* d_ws, size_t ws_size,
                              hipStream_t stream) {
    (void)in_sizes; (void)n_in; (void)out_size; (void)d_ws; (void)ws_size;
    const float* first  = (const float*)d_in[0];
    const float* second = (const float*)d_in[1];
    float* out = (float*)d_out;

    constexpr int B = 8, C = 256, H = 96, W = 160;
    dim3 grid(W / 16, H, B);   // (10, 96, 8)
    dim3 block(9 * 32);        // 9 waves: one per dy
    corr_wmma_kernel<C, H, W><<<grid, block, 0, stream>>>(first, second, out);
}
